// STICKYLlamaAttention_68496138436911
// MI455X (gfx1250) — compile-verified, matrix operand order checked by reading, not run on hardware
//
#include <hip/hip_runtime.h>

// ---------------------------------------------------------------------------
// CDNA5 / gfx1250: wave32, WMMA 16x16x32 bf16 (f32 accumulate).
// All WMMA operands are kept K-contiguous in memory so every fragment load is
// 2x b128 (no scalar ds_load_u16 gathers). GEMM tiles are staged to LDS with
// gfx1250 async-to-LDS DMA (ASYNCcnt) when available.
// ---------------------------------------------------------------------------
#ifndef USE_ASYNC_LDS
#define USE_ASYNC_LDS 1
#endif

typedef __attribute__((ext_vector_type(16))) __bf16 v16bf;
typedef __attribute__((ext_vector_type(8)))  __bf16 bf16x8;
typedef __attribute__((ext_vector_type(8)))  float  v8f;

// Element e of a v16bf A/B fragment maps to K index (ISA 7.12.2, 16-bit 16x32):
// lanes 0-15: VGPR0-3 -> K0..7, VGPR4-7 -> K16..23 ; lanes 16-31: +8
__device__ __forceinline__ int kmap(int e, int hi) {
  return (e < 8 ? e : e + 8) + hi * 8;
}

__device__ __forceinline__ v8f zero8() {
  v8f z;
#pragma unroll
  for (int i = 0; i < 8; ++i) z[i] = 0.0f;
  return z;
}

// Async global->LDS copy of one 16-byte chunk (VGLOBAL encoding, ASYNCcnt).
__device__ __forceinline__ void async_copy16(const __bf16* g, __bf16* l) {
#if USE_ASYNC_LDS
  unsigned lds = (unsigned)(uintptr_t)l;  // LDS byte address = low 32 bits of generic ptr
  asm volatile("global_load_async_to_lds_b128 %0, %1, off"
               :: "v"(lds), "v"(g) : "memory");
#else
  *(bf16x8*)l = *(const bf16x8*)g;
#endif
}

__device__ __forceinline__ void async_wait() {
#if USE_ASYNC_LDS
  asm volatile("s_wait_asynccnt 0x0" ::: "memory");
#endif
}

// ---------------------------------------------------------------------------
// Elementwise f32 -> bf16 convert
// ---------------------------------------------------------------------------
__global__ void f32_to_bf16_k(const float* __restrict__ in, __bf16* __restrict__ out, int n) {
  int i = blockIdx.x * blockDim.x + threadIdx.x;
  if (i < n) out[i] = (__bf16)in[i];
}

// f32 [R,C] -> bf16 transposed [C,R], tiled through LDS (coalesced both sides)
__global__ __launch_bounds__(256) void f32_to_bf16_tr_k(
    const float* __restrict__ in, __bf16* __restrict__ out, int R, int C)
{
  __shared__ float tile[32][33];
  const int c0 = blockIdx.x * 32, r0 = blockIdx.y * 32;
  const int tx = threadIdx.x & 31, ty = threadIdx.x >> 5;   // 32 x 8
#pragma unroll
  for (int i = 0; i < 4; ++i)
    tile[ty + i * 8][tx] = in[(size_t)(r0 + ty + i * 8) * C + c0 + tx];
  __syncthreads();
#pragma unroll
  for (int i = 0; i < 4; ++i)
    out[(size_t)(c0 + ty + i * 8) * R + r0 + tx] = (__bf16)tile[tx][ty + i * 8];
}

__global__ void zero_f32_k(float* __restrict__ p, int n) {
  int i = blockIdx.x * blockDim.x + threadIdx.x;
  if (i < n) p[i] = 0.0f;
}

// ---------------------------------------------------------------------------
// Tiled GEMM: C[M,N](f32) = A[M,K](bf16 row-major) * Bt[N,K](bf16, = B^T)
// Block 64x128, 8 waves. Both LDS tiles are K-major -> every fragment read is
// 2x ds_load_b128. Tiles staged via async-to-LDS DMA.
// ---------------------------------------------------------------------------
__global__ __launch_bounds__(256) void gemm_bf16_f32(
    const __bf16* __restrict__ A, const __bf16* __restrict__ Bt,
    float* __restrict__ C, int M, int N, int K)
{
  __shared__ __bf16 sA[64][32];    // 4 KB, K-major
  __shared__ __bf16 sB[128][32];   // 8 KB, K-major (rows = N)

  const int bm = blockIdx.y * 64;
  const int bn = blockIdx.x * 128;
  const int t    = threadIdx.x;
  const int lane = t & 31;
  const int wv   = t >> 5;
  const int hi   = lane >> 4;
  const int idx  = lane & 15;
  const int wm   = (wv & 3) * 16;
  const int wn   = (wv >> 2) * 64;

  v8f acc[4];
#pragma unroll
  for (int i = 0; i < 4; ++i) acc[i] = zero8();

  const int ar = t >> 2;          // 0..63   (A: 8 elems/thread)
  const int ac = (t & 3) * 8;
  const int br = t >> 1;          // 0..127  (Bt: 16 elems/thread)
  const int bo = (t & 1) * 16;

  for (int k0 = 0; k0 < K; k0 += 32) {
    __syncthreads();   // previous compute done before overwriting tiles
    async_copy16(&A [(size_t)(bm + ar) * K + k0 + ac],     &sA[ar][ac]);
    async_copy16(&Bt[(size_t)(bn + br) * K + k0 + bo],     &sB[br][bo]);
    async_copy16(&Bt[(size_t)(bn + br) * K + k0 + bo + 8], &sB[br][bo + 8]);
#if !USE_ASYNC_LDS
    if (k0 + 32 < K) {
      __builtin_prefetch(&A [(size_t)(bm + ar) * K + k0 + 32 + ac], 0, 0);
      __builtin_prefetch(&Bt[(size_t)(bn + br) * K + k0 + 32 + bo], 0, 0);
    }
#endif
    async_wait();      // wave's own DMA complete
    __syncthreads();   // all waves' tiles visible

    v16bf af;
#pragma unroll
    for (int e = 0; e < 16; ++e) af[e] = sA[wm + idx][kmap(e, hi)];

#pragma unroll
    for (int nt = 0; nt < 4; ++nt) {
      v16bf bfv;
#pragma unroll
      for (int e = 0; e < 16; ++e) bfv[e] = sB[wn + nt * 16 + idx][kmap(e, hi)];
      acc[nt] = __builtin_amdgcn_wmma_f32_16x16x32_bf16(
          false, af, false, bfv, (short)0, acc[nt], false, false);
    }
  }

#pragma unroll
  for (int nt = 0; nt < 4; ++nt)
#pragma unroll
    for (int r = 0; r < 8; ++r)
      C[(size_t)(bm + wm + r + 8 * hi) * N + bn + wn + nt * 16 + idx] = acc[nt][r];
}

// ---------------------------------------------------------------------------
// RoPE (f32) + repack:
//   Qf[s,4096] -> Qr[h][s][128] bf16 (K-major for QK^T A-frags)
//   Kf[s,1024] -> Kr[kh][s][128] bf16 (K-major for QK^T B-frags)
//   Vf[s,1024] -> Vt[kh][128][2048] bf16 TRANSPOSED (key-contiguous for PV B-frags)
// ---------------------------------------------------------------------------
__global__ __launch_bounds__(256) void rope_pack_k(
    const float* __restrict__ Qf, const float* __restrict__ Kf, const float* __restrict__ Vf,
    __bf16* __restrict__ Qr, __bf16* __restrict__ Kr, __bf16* __restrict__ Vt,
    const int* __restrict__ pos)
{
  const int s = blockIdx.y;
  const int c = blockIdx.x * blockDim.x + threadIdx.x;   // 0..4095
  const int d  = c & 127;
  const int hh = c >> 7;
  const float p = (float)pos[s];
  // inv_freq = 10000^(-2i/128), i = d mod 64
  const float inv = __expf(-(float)(2 * (d & 63)) * (9.210340371976184f / 128.0f));
  const float fr = p * inv;
  const float cs = __cosf(fr), sn = __sinf(fr);

  const float q  = Qf[(size_t)s * 4096 + c];
  const float qr = (d < 64) ? -Qf[(size_t)s * 4096 + c + 64] : Qf[(size_t)s * 4096 + c - 64];
  Qr[((size_t)hh * 2048 + s) * 128 + d] = (__bf16)(q * cs + qr * sn);

  if (c < 1024) {
    const float kv = Kf[(size_t)s * 1024 + c];
    const float kr = (d < 64) ? -Kf[(size_t)s * 1024 + c + 64] : Kf[(size_t)s * 1024 + c - 64];
    Kr[((size_t)hh * 2048 + s) * 128 + d] = (__bf16)(kv * cs + kr * sn);
    Vt[((size_t)hh * 128 + d) * 2048 + s] = (__bf16)Vf[(size_t)s * 1024 + c];
  }
}

// ---------------------------------------------------------------------------
// QK^T tiles: s0 = Q(16x128) * K[kb..kb+15]^T, s1 = keys kb+16..kb+31
// (K rows are hd-contiguous -> vectorized global loads)
// ---------------------------------------------------------------------------
__device__ __forceinline__ void qk_tiles(const v16bf* qf, const __bf16* __restrict__ Kh,
                                         int kb, int hi, int idx, v8f& s0, v8f& s1)
{
  s0 = zero8();
  s1 = zero8();
#pragma unroll
  for (int kc = 0; kc < 4; ++kc) {
    v16bf k0f, k1f;
#pragma unroll
    for (int e = 0; e < 16; ++e) {
      const int kk = kc * 32 + kmap(e, hi);
      k0f[e] = Kh[(size_t)(kb + idx) * 128 + kk];
      k1f[e] = Kh[(size_t)(kb + 16 + idx) * 128 + kk];
    }
    s0 = __builtin_amdgcn_wmma_f32_16x16x32_bf16(false, qf[kc], false, k0f, (short)0, s0, false, false);
    s1 = __builtin_amdgcn_wmma_f32_16x16x32_bf16(false, qf[kc], false, k1f, (short)0, s1, false, false);
  }
}

// ---------------------------------------------------------------------------
// Causal flash attention + eviction-score pass.
// 1 wave per (head, 16-query block); 4 independent waves per workgroup.
// ---------------------------------------------------------------------------
__global__ __launch_bounds__(128) void attn_k(
    const __bf16* __restrict__ Qr, const __bf16* __restrict__ Kr, const __bf16* __restrict__ Vt,
    __bf16* __restrict__ attnb, float* __restrict__ scores)
{
  __shared__ __bf16 sP[4][16][32];   // per-wave P staging (C-layout -> A-frag layout)

  const int lane = threadIdx.x & 31;
  const int wv   = threadIdx.x >> 5;
  const int gw   = blockIdx.x * 4 + wv;
  const int h    = gw >> 7;       // 0..31
  const int qb   = gw & 127;      // 0..127
  const int kvh  = h >> 2;
  const int q0   = qb * 16;
  const int hi   = lane >> 4;
  const int idx  = lane & 15;
  const float scale = 0.08838834764831845f;  // 1/sqrt(128)

  const __bf16* Qh = Qr + (size_t)h   * 2048 * 128;
  const __bf16* Kh = Kr + (size_t)kvh * 2048 * 128;
  const __bf16* Vh = Vt + (size_t)kvh * 128 * 2048;   // [hd][key]

  // Preload Q as 4 A-fragments covering HD=128
  v16bf qf[4];
#pragma unroll
  for (int kc = 0; kc < 4; ++kc)
#pragma unroll
    for (int e = 0; e < 16; ++e)
      qf[kc][e] = Qh[(size_t)(q0 + idx) * 128 + kc * 32 + kmap(e, hi)];

  v8f o[8];
#pragma unroll
  for (int i = 0; i < 8; ++i) o[i] = zero8();
  float mrow[8], lrow[8];
#pragma unroll
  for (int r = 0; r < 8; ++r) { mrow[r] = -1e30f; lrow[r] = 0.0f; }

  const int kb_end = ((q0 + 15) >> 5) << 5;  // last 32-key block start (causal)

  for (int kb = 0; kb <= kb_end; kb += 32) {
    v8f s0, s1;
    qk_tiles(qf, Kh, kb, hi, idx, s0, s1);

    float p0[8], p1[8], alpha[8];
#pragma unroll
    for (int r = 0; r < 8; ++r) {
      const int row = q0 + r + 8 * hi;
      float v0 = (kb + idx      > row) ? -1e30f : s0[r] * scale;
      float v1 = (kb + 16 + idx > row) ? -1e30f : s1[r] * scale;
      float m = fmaxf(v0, v1);
#pragma unroll
      for (int off = 1; off < 16; off <<= 1) m = fmaxf(m, __shfl_xor(m, off, 32));
      const float mn = fmaxf(mrow[r], m);
      const float al = __expf(mrow[r] - mn);
      const float e0 = __expf(v0 - mn);
      const float e1 = __expf(v1 - mn);
      float rs = e0 + e1;
#pragma unroll
      for (int off = 1; off < 16; off <<= 1) rs += __shfl_xor(rs, off, 32);
      lrow[r] = lrow[r] * al + rs;
      mrow[r] = mn;
      alpha[r] = al;
      p0[r] = e0;
      p1[r] = e1;
    }
#pragma unroll
    for (int nt = 0; nt < 8; ++nt)
#pragma unroll
      for (int r = 0; r < 8; ++r) o[nt][r] *= alpha[r];

    // C-layout P -> A-frag layout via per-wave LDS tile (DS ops are in-order)
#pragma unroll
    for (int r = 0; r < 8; ++r) {
      sP[wv][r + 8 * hi][idx]      = (__bf16)p0[r];
      sP[wv][r + 8 * hi][16 + idx] = (__bf16)p1[r];
    }
    v16bf pf;
#pragma unroll
    for (int e = 0; e < 16; ++e) pf[e] = sP[wv][idx][kmap(e, hi)];

    // O += P(16x32) * V(32x128); Vt is key-contiguous -> vectorized loads
#pragma unroll
    for (int nt = 0; nt < 8; ++nt) {
      v16bf vf;
#pragma unroll
      for (int e = 0; e < 16; ++e)
        vf[e] = Vh[(size_t)(nt * 16 + idx) * 2048 + kb + kmap(e, hi)];
      o[nt] = __builtin_amdgcn_wmma_f32_16x16x32_bf16(
          false, pf, false, vf, (short)0, o[nt], false, false);
    }
  }

  float linv[8];
#pragma unroll
  for (int r = 0; r < 8; ++r) linv[r] = 1.0f / lrow[r];

  // attnb[s, h*128 + d] bf16 (K-major A operand of the output projection)
#pragma unroll
  for (int nt = 0; nt < 8; ++nt)
#pragma unroll
    for (int r = 0; r < 8; ++r)
      attnb[(size_t)(q0 + r + 8 * hi) * 4096 + h * 128 + nt * 16 + idx] =
          (__bf16)(o[nt][r] * linv[r]);

  // Pass 2: accumulated eviction scores = sum_q softmax(q,k), per KV head.
  for (int kb = 0; kb <= kb_end; kb += 32) {
    v8f s0, s1;
    qk_tiles(qf, Kh, kb, hi, idx, s0, s1);
    float cs0 = 0.0f, cs1 = 0.0f;
#pragma unroll
    for (int r = 0; r < 8; ++r) {
      const int row = q0 + r + 8 * hi;
      if (kb + idx      <= row) cs0 += __expf(s0[r] * scale - mrow[r]) * linv[r];
      if (kb + 16 + idx <= row) cs1 += __expf(s1[r] * scale - mrow[r]) * linv[r];
    }
    cs0 += __shfl_xor(cs0, 16, 32);
    cs1 += __shfl_xor(cs1, 16, 32);
    if (hi == 0) {
      atomicAdd(&scores[kvh * 2048 + kb + idx],      cs0);
      atomicAdd(&scores[kvh * 2048 + kb + 16 + idx], cs1);
    }
  }
}

// ---------------------------------------------------------------------------
// Host launcher
// ---------------------------------------------------------------------------
extern "C" void kernel_launch(void* const* d_in, const int* in_sizes, int n_in,
                              void* d_out, int out_size, void* d_ws, size_t ws_size,
                              hipStream_t stream)
{
  (void)in_sizes; (void)n_in; (void)out_size; (void)ws_size;

  const float* hs  = (const float*)d_in[0];   // [2048,4096]
  const float* wq  = (const float*)d_in[1];   // [4096,4096]
  const float* wk  = (const float*)d_in[2];   // [4096,1024]
  const float* wv  = (const float*)d_in[3];   // [4096,1024]
  const float* wo  = (const float*)d_in[4];   // [4096,4096]
  const int*   pos = (const int*)d_in[5];     // [2048]
  float* out = (float*)d_out;                 // [2048*4096] attn ++ [8*2048] scores

  char* ws = (char*)d_ws;
  const size_t MB = 1024ull * 1024ull;
  __bf16* Xb  = (__bf16*)(ws + 0);        // 16 MB  X   [2048,4096]
  __bf16* Wqt = (__bf16*)(ws + 16 * MB);  // 32 MB  Wq^T [4096,4096]
  __bf16* Wkt = (__bf16*)(ws + 48 * MB);  //  8 MB  Wk^T [1024,4096]
  __bf16* Wvt = (__bf16*)(ws + 56 * MB);  //  8 MB  Wv^T [1024,4096]
  __bf16* Wot = (__bf16*)(ws + 64 * MB);  // 32 MB  Wo^T [4096,4096]
  float*  Qf  = (float*)(ws + 96 * MB);   // 32 MB  (region reused as attn bf16 later)
  float*  Kf  = (float*)(ws + 128 * MB);  //  8 MB
  float*  Vf  = (float*)(ws + 136 * MB);  //  8 MB
  __bf16* Qr  = (__bf16*)(ws + 144 * MB); // 16 MB  [32][2048][128]
  __bf16* Kr  = (__bf16*)(ws + 160 * MB); //  4 MB  [8][2048][128]
  __bf16* Vt  = (__bf16*)(ws + 164 * MB); //  4 MB  [8][128][2048]  (total 168 MB)
  __bf16* Ab  = (__bf16*)(ws + 96 * MB);  // reuse dead Qf region

  // 1) f32 -> bf16 conversions; weights transposed to [N,K] for K-major B-frags
  f32_to_bf16_k<<<(2048 * 4096) / 256, 256, 0, stream>>>(hs, Xb, 2048 * 4096);
  f32_to_bf16_tr_k<<<dim3(4096 / 32, 4096 / 32), 256, 0, stream>>>(wq, Wqt, 4096, 4096);
  f32_to_bf16_tr_k<<<dim3(1024 / 32, 4096 / 32), 256, 0, stream>>>(wk, Wkt, 4096, 1024);
  f32_to_bf16_tr_k<<<dim3(1024 / 32, 4096 / 32), 256, 0, stream>>>(wv, Wvt, 4096, 1024);
  f32_to_bf16_tr_k<<<dim3(4096 / 32, 4096 / 32), 256, 0, stream>>>(wo, Wot, 4096, 4096);

  // 2) QKV projections (WMMA GEMMs, async tile staging)
  gemm_bf16_f32<<<dim3(4096 / 128, 2048 / 64), 256, 0, stream>>>(Xb, Wqt, Qf, 2048, 4096, 4096);
  gemm_bf16_f32<<<dim3(1024 / 128, 2048 / 64), 256, 0, stream>>>(Xb, Wkt, Kf, 2048, 1024, 4096);
  gemm_bf16_f32<<<dim3(1024 / 128, 2048 / 64), 256, 0, stream>>>(Xb, Wvt, Vf, 2048, 1024, 4096);

  // 3) RoPE + head-major bf16 repack (V transposed to key-contiguous)
  rope_pack_k<<<dim3(16, 2048), 256, 0, stream>>>(Qf, Kf, Vf, Qr, Kr, Vt, pos);

  // 4) scores = 0, then flash attention + score accumulation
  float* scores = out + (size_t)2048 * 4096;
  zero_f32_k<<<(8 * 2048) / 256, 256, 0, stream>>>(scores, 8 * 2048);
  attn_k<<<dim3(32 * 128 / 4), 128, 0, stream>>>(Qr, Kr, Vt, Ab, scores);

  // 5) Output projection -> d_out (f32)
  gemm_bf16_f32<<<dim3(4096 / 128, 2048 / 64), 256, 0, stream>>>(Ab, Wot, out, 2048, 4096, 4096);
}
